// CrossAttention_59021440582234
// MI455X (gfx1250) — matrix operand order repaired and checked by reading, not compiled, not run on hardware
//
#include <hip/hip_runtime.h>
#include <hip/hip_bf16.h>

#define BB     2
#define MM     1024
#define NN     2048
#define DM     1024
#define NH     16
#define NHKV   4
#define DH     64
#define GRP    (NH / NHKV)
#define MAXREL (MM + NN)
#define SPLIT  4                  // key-dim splits (one per wave in an attention block)
#define KSPAN  (NN / SPLIT)
#define LOGTHETA 9.210340371976184f   // ln(10000)

typedef __attribute__((ext_vector_type(16))) __bf16 bfx16;
typedef __attribute__((ext_vector_type(8)))  float  fx8;

#define WMMA_BF16(a, b, c) \
    __builtin_amdgcn_wmma_f32_16x16x32_bf16(false, (a), false, (b), (short)0, (c), false, false)

// ---------- fragment loaders (layouts per CDNA5 ISA) ----------

__device__ __forceinline__ bfx16 ld_pair(const __bf16* p0, const __bf16* p1) {
    union { bfx16 v; uint4 u[2]; } t;
    t.u[0] = *reinterpret_cast<const uint4*>(p0);
    t.u[1] = *reinterpret_cast<const uint4*>(p1);
    return t.v;
}

// A (16x32 bf16), row-major [M x K]: lane m = lane&15, half = lane>>4.
// VGPR j: K = (j>>2)*16 + half*8 + (j&3)*2 (+1)  -> two contiguous 8-elem runs.
__device__ __forceinline__ bfx16 ld_afrag(const __bf16* A, int lda, int m0, int k0, int lane) {
    int m = lane & 15, half = lane >> 4;
    const __bf16* p = A + (size_t)(m0 + m) * lda + k0 + half * 8;
    return ld_pair(p, p + 16);
}

// B (32x16 bf16) given B^T stored row-major [N x K]: lane n = lane&15,
// half = lane>>4; K = half*8 + [0,8) and 16 + half*8 + [0,8) (same as A).
__device__ __forceinline__ bfx16 ld_bfragT(const __bf16* Bt, int ldk, int n0, int k0, int lane) {
    int n = lane & 15, kh = (lane >> 4) * 8;
    const __bf16* p = Bt + (size_t)(n0 + n) * ldk + k0 + kh;
    return ld_pair(p, p + 16);
}

__device__ __forceinline__ fx8 fx8_zero() {
    fx8 z = {0.f, 0.f, 0.f, 0.f, 0.f, 0.f, 0.f, 0.f};
    return z;
}

// ---------- conversion / table kernels ----------

__global__ void k_cvt(const float* __restrict__ s, __bf16* __restrict__ d, int n) {
    int i = blockIdx.x * blockDim.x + threadIdx.x;
    if (i < n) d[i] = (__bf16)s[i];
}

// src [K][N] f32  ->  dst [N][K] bf16  (transpose so B-frags are contiguous)
__global__ void k_cvt_t(const float* __restrict__ s, __bf16* __restrict__ d, int K, int N) {
    int i = blockIdx.x * blockDim.x + threadIdx.x;
    if (i < K * N) {
        int k = i / N, n = i - k * N;
        d[(size_t)n * K + k] = (__bf16)s[i];
    }
}

// RoPE tables: cos/sin(pos * theta^(-2*(d/2)/64)) for pos in [0,NN), d in [0,64)
__global__ void k_ropetab(float* __restrict__ cosT, float* __restrict__ sinT) {
    int i = blockIdx.x * blockDim.x + threadIdx.x;
    if (i < NN * DH) {
        int pos = i >> 6, d = i & 63;
        float f = __expf(-(float)(d >> 1) * (LOGTHETA / 32.f));
        float a = (float)pos * f;
        cosT[i] = __cosf(a);
        sinT[i] = __sinf(a);
    }
}

// ---------- Q projection + RoPE ----------
// grid (MM/64, NH, BB), 128 threads = 4 waves, each wave: 16 rows x 64 cols (one head)
__global__ __launch_bounds__(128) void k_qproj(const __bf16* __restrict__ qb,
                                               const __bf16* __restrict__ WqT,
                                               const float* __restrict__ cosT,
                                               const float* __restrict__ sinT,
                                               __bf16* __restrict__ Qr) {
    int lane = threadIdx.x & 31, w = threadIdx.x >> 5;
    int m0 = (blockIdx.x * 4 + w) * 16;
    int h = blockIdx.y, b = blockIdx.z;
    const __bf16* A = qb + (size_t)b * MM * DM;

    fx8 c[4];
    for (int t = 0; t < 4; ++t) c[t] = fx8_zero();

    for (int k0 = 0; k0 < DM; k0 += 32) {
        bfx16 a = ld_afrag(A, DM, m0, k0, lane);
        for (int t = 0; t < 4; ++t) {
            bfx16 bb = ld_bfragT(WqT, DM, h * DH + t * 16, k0, lane);
            c[t] = WMMA_BF16(a, bb, c[t]);
        }
    }

    int n = lane & 15, half = lane >> 4;
    for (int pi = 0; pi < 2; ++pi) {            // tile pairs (0,2) and (1,3): d and d+32
        int ta = pi, tb = pi + 2;
        int dlo = ta * 16 + n;
        for (int r = 0; r < 8; ++r) {
            int pos = m0 + r + 8 * half;
            int ti = pos * DH + dlo;
            float c1 = cosT[ti], s1 = sinT[ti];
            float c2 = cosT[ti + 32], s2 = sinT[ti + 32];
            float x1 = c[ta][r], x2 = c[tb][r];
            float olo = x1 * c1 - x2 * s1;
            float ohi = x1 * s2 + x2 * c2;
            size_t base = ((size_t)(b * NH + h) * MM + pos) * DH;
            Qr[base + dlo]      = (__bf16)olo;
            Qr[base + dlo + 32] = (__bf16)ohi;
        }
    }
}

// ---------- K/V projection: K gets RoPE, stored [N][D]; V stored transposed [D][N] ----------
// grid (NN/64, NHKV, BB), 128 threads
__global__ __launch_bounds__(128) void k_kvproj(const __bf16* __restrict__ kvb,
                                                const __bf16* __restrict__ WkT,
                                                const __bf16* __restrict__ WvT,
                                                const float* __restrict__ cosT,
                                                const float* __restrict__ sinT,
                                                __bf16* __restrict__ Kr,
                                                __bf16* __restrict__ Vt) {
    int lane = threadIdx.x & 31, w = threadIdx.x >> 5;
    int n0 = (blockIdx.x * 4 + w) * 16;
    int h = blockIdx.y, b = blockIdx.z;
    const __bf16* A = kvb + (size_t)b * NN * DM;

    fx8 ck[4], cv[4];
    for (int t = 0; t < 4; ++t) { ck[t] = fx8_zero(); cv[t] = fx8_zero(); }

    for (int k0 = 0; k0 < DM; k0 += 32) {
        bfx16 a = ld_afrag(A, DM, n0, k0, lane);
        for (int t = 0; t < 4; ++t) {
            bfx16 bk = ld_bfragT(WkT, DM, h * DH + t * 16, k0, lane);
            ck[t] = WMMA_BF16(a, bk, ck[t]);
            bfx16 bv = ld_bfragT(WvT, DM, h * DH + t * 16, k0, lane);
            cv[t] = WMMA_BF16(a, bv, cv[t]);
        }
    }

    int n = lane & 15, half = lane >> 4;
    // K: RoPE, store [B][HKV][N][D]
    for (int pi = 0; pi < 2; ++pi) {
        int ta = pi, tb = pi + 2;
        int dlo = ta * 16 + n;
        for (int r = 0; r < 8; ++r) {
            int pos = n0 + r + 8 * half;
            int ti = pos * DH + dlo;
            float c1 = cosT[ti], s1 = sinT[ti];
            float c2 = cosT[ti + 32], s2 = sinT[ti + 32];
            float x1 = ck[ta][r], x2 = ck[tb][r];
            float olo = x1 * c1 - x2 * s1;
            float ohi = x1 * s2 + x2 * c2;
            size_t base = ((size_t)(b * NHKV + h) * NN + pos) * DH;
            Kr[base + dlo]      = (__bf16)olo;
            Kr[base + dlo + 32] = (__bf16)ohi;
        }
    }
    // V: store transposed [B][HKV][D][N]
    for (int t = 0; t < 4; ++t)
        for (int r = 0; r < 8; ++r) {
            int pos = n0 + r + 8 * half;
            int d = t * 16 + n;
            Vt[((size_t)(b * NHKV + h) * DH + d) * NN + pos] = (__bf16)cv[t][r];
        }
}

// ---------- fused flash attention, key-dim split across the 4 waves of a block ----------
// grid (MM/16, NH, BB), 128 threads = 4 waves. Wave w handles keys [w*KSPAN, (w+1)*KSPAN)
// of the SAME 16-query tile; partial flash states combined in LDS at the end.
__global__ __launch_bounds__(128) void k_attn(const __bf16* __restrict__ Qr,
                                              const __bf16* __restrict__ Kr,
                                              const __bf16* __restrict__ Vt,
                                              const float* __restrict__ bias,
                                              const unsigned char* __restrict__ mask,
                                              __bf16* __restrict__ O) {
    __shared__ __align__(16) __bf16 smP[SPLIT][16 * 32];   // per-wave P staging tile
    __shared__ float sAcc[SPLIT][16 * DH];                 // per-wave un-normalized outputs
    __shared__ float sM[SPLIT][16], sL[SPLIT][16], sW[SPLIT][16];

    int lane = threadIdx.x & 31, w = threadIdx.x >> 5;
    __bf16* smw = &smP[w][0];
    int m0 = blockIdx.x * 16;
    int h = blockIdx.y, b = blockIdx.z;
    int hk = h / GRP;

    const __bf16* Qh = Qr + (size_t)(b * NH + h) * MM * DH;
    const __bf16* Kh = Kr + (size_t)(b * NHKV + hk) * NN * DH;
    const __bf16* Vh = Vt + (size_t)(b * NHKV + hk) * DH * NN;
    const float* bh = bias + (size_t)h * MAXREL;
    const unsigned char* mk = mask + (size_t)b * NN;

    int n = lane & 15, half = lane >> 4;

    bfx16 qa0 = ld_afrag(Qh, DH, m0, 0, lane);
    bfx16 qa1 = ld_afrag(Qh, DH, m0, 32, lane);

    fx8 acc[4];
    for (int t = 0; t < 4; ++t) acc[t] = fx8_zero();
    float Mst[8], Lst[8];
    for (int r = 0; r < 8; ++r) { Mst[r] = -3.0e38f; Lst[r] = 0.f; }

    int jbeg = w * KSPAN, jend = jbeg + KSPAN;
    for (int j0 = jbeg; j0 < jend; j0 += 32) {
        fx8 s0 = fx8_zero(), s1 = fx8_zero();
        {
            bfx16 kb = ld_bfragT(Kh, DH, j0, 0, lane);
            s0 = WMMA_BF16(qa0, kb, s0);
            kb = ld_bfragT(Kh, DH, j0, 32, lane);
            s0 = WMMA_BF16(qa1, kb, s0);
            kb = ld_bfragT(Kh, DH, j0 + 16, 0, lane);
            s1 = WMMA_BF16(qa0, kb, s1);
            kb = ld_bfragT(Kh, DH, j0 + 16, 32, lane);
            s1 = WMMA_BF16(qa1, kb, s1);
        }
        int key0 = j0 + n, key1 = j0 + 16 + n;
        float msk0 = mk[key0] ? 0.f : -1.0e9f;
        float msk1 = mk[key1] ? 0.f : -1.0e9f;

        for (int r = 0; r < 8; ++r) {
            int mrow = m0 + r + 8 * half;
            int d0 = mrow - key0; if (d0 < 0) d0 = -d0;
            int d1 = mrow - key1; if (d1 < 0) d1 = -d1;
            float v0 = s0[r] * 0.125f + bh[d0] + msk0;
            float v1 = s1[r] * 0.125f + bh[d1] + msk1;

            float tmax = fmaxf(v0, v1);
            for (int off = 1; off < 16; off <<= 1)
                tmax = fmaxf(tmax, __shfl_xor(tmax, off, 16));
            float newM = fmaxf(Mst[r], tmax);
            float corr = __expf(Mst[r] - newM);
            Mst[r] = newM;
            float p0 = __expf(v0 - newM);
            float p1 = __expf(v1 - newM);
            float ps = p0 + p1;
            for (int off = 1; off < 16; off <<= 1)
                ps += __shfl_xor(ps, off, 16);
            Lst[r] = Lst[r] * corr + ps;
            for (int t = 0; t < 4; ++t) acc[t][r] *= corr;

            int mr = r + 8 * half;                 // C-layout -> LDS [16][32] row-major
            smw[mr * 32 + n]      = (__bf16)p0;
            smw[mr * 32 + 16 + n] = (__bf16)p1;
        }

        // re-read P in A-fragment layout (same-wave DS ops are in-order)
        const __bf16* pp = smw + (size_t)(lane & 15) * 32 + half * 8;
        bfx16 pa = ld_pair(pp, pp + 16);

        for (int t = 0; t < 4; ++t) {
            bfx16 vb = ld_bfragT(Vh, NN, t * 16, j0, lane);
            acc[t] = WMMA_BF16(pa, vb, acc[t]);
        }
    }

    // ---- combine the 4 key-split partial flash states in LDS ----
    if (n == 0) {
        for (int r = 0; r < 8; ++r) {
            int mr = r + 8 * half;
            sM[w][mr] = Mst[r];
            sL[w][mr] = Lst[r];
        }
    }
    for (int t = 0; t < 4; ++t)
        for (int r = 0; r < 8; ++r)
            sAcc[w][(r + 8 * half) * DH + t * 16 + n] = acc[t][r];
    __syncthreads();

    if (threadIdx.x < 16) {
        int mr = threadIdx.x;
        float g = sM[0][mr];
        for (int s = 1; s < SPLIT; ++s) g = fmaxf(g, sM[s][mr]);
        float ws[SPLIT], Lc = 0.f;
        for (int s = 0; s < SPLIT; ++s) {
            ws[s] = __expf(sM[s][mr] - g);
            Lc += ws[s] * sL[s][mr];
        }
        float inv = 1.f / Lc;
        for (int s = 0; s < SPLIT; ++s) sW[s][mr] = ws[s] * inv;
    }
    __syncthreads();

    for (int e = threadIdx.x; e < 16 * DH; e += 128) {
        int mr = e >> 6, d = e & 63;
        float v = 0.f;
        for (int s = 0; s < SPLIT; ++s) v += sAcc[s][e] * sW[s][mr];
        O[(size_t)(b * MM + m0 + mr) * DM + h * DH + d] = (__bf16)v;
    }
}

// ---------- output projection: out = O @ Wo (fp32 out) ----------
// rows = BB*MM = 2048. grid (2048/64, DM/64), 128 threads
__global__ __launch_bounds__(128) void k_out(const __bf16* __restrict__ Ob,
                                             const __bf16* __restrict__ WoT,
                                             float* __restrict__ out) {
    int lane = threadIdx.x & 31, w = threadIdx.x >> 5;
    int row0 = (blockIdx.x * 4 + w) * 16;
    int col0 = blockIdx.y * 64;

    fx8 c[4];
    for (int t = 0; t < 4; ++t) c[t] = fx8_zero();

    for (int k0 = 0; k0 < DM; k0 += 32) {
        bfx16 a = ld_afrag(Ob, DM, row0, k0, lane);
        for (int t = 0; t < 4; ++t) {
            bfx16 bb = ld_bfragT(WoT, DM, col0 + t * 16, k0, lane);
            c[t] = WMMA_BF16(a, bb, c[t]);
        }
    }
    int n = lane & 15, half = lane >> 4;
    for (int t = 0; t < 4; ++t)
        for (int r = 0; r < 8; ++r)
            out[(size_t)(row0 + r + 8 * half) * DM + col0 + t * 16 + n] = c[t][r];
}

// ---------- host launcher ----------

extern "C" void kernel_launch(void* const* d_in, const int* in_sizes, int n_in,
                              void* d_out, int out_size, void* d_ws, size_t ws_size,
                              hipStream_t stream) {
    const float* q    = (const float*)d_in[0];
    const float* kv   = (const float*)d_in[1];
    const float* Wq   = (const float*)d_in[2];
    const float* Wk   = (const float*)d_in[3];
    const float* Wv   = (const float*)d_in[4];
    const float* Wo   = (const float*)d_in[5];
    const float* bias = (const float*)d_in[6];
    const unsigned char* mask = (const unsigned char*)d_in[7];
    float* out = (float*)d_out;

    char* w = (char*)d_ws;
    size_t off = 0;
    auto takeB = [&](size_t bytes) { char* p = w + off; off += (bytes + 255) & ~(size_t)255; return p; };

    __bf16* qb   = (__bf16*)takeB((size_t)BB * MM * DM * 2);
    __bf16* kvb  = (__bf16*)takeB((size_t)BB * NN * DM * 2);
    __bf16* WqT  = (__bf16*)takeB((size_t)DM * (NH * DH) * 2);
    __bf16* WkT  = (__bf16*)takeB((size_t)DM * (NHKV * DH) * 2);
    __bf16* WvT  = (__bf16*)takeB((size_t)DM * (NHKV * DH) * 2);
    __bf16* WoT  = (__bf16*)takeB((size_t)(NH * DH) * DM * 2);
    __bf16* Qr   = (__bf16*)takeB((size_t)BB * NH * MM * DH * 2);
    __bf16* Kr   = (__bf16*)takeB((size_t)BB * NHKV * NN * DH * 2);
    __bf16* Vt   = (__bf16*)takeB((size_t)BB * NHKV * DH * NN * 2);
    __bf16* Ob   = (__bf16*)takeB((size_t)BB * MM * DM * 2);
    float*  cosT = (float*)takeB((size_t)NN * DH * 4);
    float*  sinT = (float*)takeB((size_t)NN * DH * 4);

    int nq = BB * MM * DM, nkv = BB * NN * DM;
    k_cvt<<<(nq + 255) / 256, 256, 0, stream>>>(q, qb, nq);
    k_cvt<<<(nkv + 255) / 256, 256, 0, stream>>>(kv, kvb, nkv);
    k_cvt_t<<<(DM * NH * DH + 255) / 256, 256, 0, stream>>>(Wq, WqT, DM, NH * DH);
    k_cvt_t<<<(DM * NHKV * DH + 255) / 256, 256, 0, stream>>>(Wk, WkT, DM, NHKV * DH);
    k_cvt_t<<<(DM * NHKV * DH + 255) / 256, 256, 0, stream>>>(Wv, WvT, DM, NHKV * DH);
    k_cvt_t<<<(NH * DH * DM + 255) / 256, 256, 0, stream>>>(Wo, WoT, NH * DH, DM);
    k_ropetab<<<(NN * DH + 255) / 256, 256, 0, stream>>>(cosT, sinT);

    k_qproj<<<dim3(MM / 64, NH, BB), 128, 0, stream>>>(qb, WqT, cosT, sinT, Qr);
    k_kvproj<<<dim3(NN / 64, NHKV, BB), 128, 0, stream>>>(kvb, WkT, WvT, cosT, sinT, Kr, Vt);
    k_attn<<<dim3(MM / 16, NH, BB), 128, 0, stream>>>(Qr, Kr, Vt, bias, mask, Ob);
    k_out<<<dim3((BB * MM) / 64, DM / 64), 128, 0, stream>>>(Ob, WoT, out);
}
